// Residual_Moe_20083267076434
// MI455X (gfx1250) — matrix operand order repaired and checked by reading, not compiled, not run on hardware
//
#include <hip/hip_runtime.h>
#include <math.h>

// ---------------- types ----------------
typedef __attribute__((ext_vector_type(16))) __bf16 v16bf;
typedef __attribute__((ext_vector_type(8)))  __bf16 bf16x8;
typedef __attribute__((ext_vector_type(8)))  float  v8f;

#define DEVINL static __device__ __forceinline__

// ---------------- problem constants ----------------
constexpr int S_   = 24;
constexpr int NB   = 2048;          // batch
constexpr int IND  = 256;
constexpr int DM   = 512;
constexpr int NH   = 8;
constexpr int HDm  = 64;
constexpr int FFD  = 2048;
constexpr int NE   = 4;
constexpr int OUTD = 64;
constexpr int SB   = S_ * NB;       // 49152 tokens
constexpr float ASCALE = 0.125f;    // 1/sqrt(64)

// ---------------- bf16 helpers ----------------
DEVINL float bf2f(__bf16 x) {
  unsigned short u = __builtin_bit_cast(unsigned short, x);
  unsigned int w = ((unsigned int)u) << 16;
  return __builtin_bit_cast(float, w);
}
DEVINL __bf16 f2bf(float f) {
  unsigned int u = __builtin_bit_cast(unsigned int, f);
  u += 0x7FFFu + ((u >> 16) & 1u);                 // round-to-nearest-even
  unsigned short s = (unsigned short)(u >> 16);
  return __builtin_bit_cast(__bf16, s);
}
DEVINL v16bf mk_frag(const __bf16* lo, const __bf16* hi) {
  bf16x8 a = *(const bf16x8*)lo;                   // ds_load_b128
  bf16x8 b = *(const bf16x8*)hi;                   // ds_load_b128
  return __builtin_shufflevector(a, b, 0,1,2,3,4,5,6,7,8,9,10,11,12,13,14,15);
}

// CDNA5 async global->LDS copy (ASYNCcnt tracked, bypasses VGPRs).
// INST_OFFSET applies to BOTH global and LDS addresses (ISA 10.7), so the
// 4 contiguous 16B chunks of one thread share a single base address pair.
#define ASYNC_LD_B128(lds, g, OFFSTR)                                        \
  asm volatile("global_load_async_to_lds_b128 %0, %1, off offset:" OFFSTR   \
               :: "v"(lds), "v"(g) : "memory")

DEVINL void async_stage64(unsigned lds, const void* g) {
  ASYNC_LD_B128(lds, g, "0");
  ASYNC_LD_B128(lds, g, "16");
  ASYNC_LD_B128(lds, g, "32");
  ASYNC_LD_B128(lds, g, "48");
}
DEVINL void wait_async0() {
  asm volatile("s_wait_asynccnt 0x0" ::: "memory");
}

// ======================================================================
// WMMA bf16 GEMM:  C[M,N] = A[M,K](bf16) @ B[K,N](bf16) [+bias][+res][relu]
// block tile 128x128x64, 8 waves, wave tile 32x64 (2x4 wmma frags, 2 k-steps).
// Double-buffered LDS; next K-tile staged with global_load_async_to_lds_b128
// while the current tile feeds v_wmma_f32_16x16x32_bf16.
// REQUIRES: M % 128 == 0, N % 128 == 0, K % 64 == 0 (true at every call site).
// ======================================================================
template<bool BIAS, bool RELU, bool RES, bool OUTBF>
__global__ __launch_bounds__(256) void k_gemm(
    const __bf16* __restrict__ A, const __bf16* __restrict__ Bw,
    const float* __restrict__ bias, const float* __restrict__ res,
    float* __restrict__ Cf, __bf16* __restrict__ Cb,
    int M, int N, int K)
{
  __shared__ __bf16 As[2][128][72];    // 64 + 8 pad; row stride 144B (16B mult)
  __shared__ __bf16 Bs[2][64][136];    // 128 + 8 pad; row stride 272B (16B mult)

  const int tid  = threadIdx.x;
  const int lane = tid & 31;
  const int wave = tid >> 5;
  const int wm   = wave & 3;        // 4 wave-rows of 32
  const int wn   = wave >> 2;       // 2 wave-cols of 64
  const long bm  = (long)blockIdx.y * 128;
  const long bn  = (long)blockIdx.x * 128;

  v8f zacc = {};
  v8f acc[2][4];
#pragma unroll
  for (int i = 0; i < 2; ++i)
#pragma unroll
    for (int j = 0; j < 4; ++j) acc[i][j] = zacc;

  // staging assignment: A row t>>1, 32-col half; B row t>>2, 32-col quarter
  const int arow  = tid >> 1;
  const int acolb = (tid & 1) << 5;
  const int brow  = tid >> 2;
  const int bcolb = (tid & 3) << 5;

  const unsigned ldsA[2] = { (unsigned)(unsigned long long)&As[0][arow][acolb],
                             (unsigned)(unsigned long long)&As[1][arow][acolb] };
  const unsigned ldsB[2] = { (unsigned)(unsigned long long)&Bs[0][brow][bcolb],
                             (unsigned)(unsigned long long)&Bs[1][brow][bcolb] };

  const __bf16* pa = A  + (bm + arow) * (long)K + acolb;       // next A chunk
  const __bf16* pb = Bw + (long)brow * N + bn + bcolb;         // next B chunk
  const long pbStep = (long)64 * N;

  async_stage64(ldsA[0], pa);  pa += 64;
  async_stage64(ldsB[0], pb);  pb += pbStep;

  int cur = 0;
  for (int k0 = 0; k0 < K; k0 += 64) {
    wait_async0();          // staged tile landed in LDS
    __syncthreads();        // visible to all waves; all done reading cur^1
    if (k0 + 64 < K) {      // overlap next stage with the math below
      async_stage64(ldsA[cur ^ 1], pa);  pa += 64;
      async_stage64(ldsB[cur ^ 1], pb);  pb += pbStep;
    }

    // ---- 2 k-substeps of 32; fragments per ISA 7.12.2 ----
#pragma unroll
    for (int kk = 0; kk < 64; kk += 32) {
      v16bf af[2], bfr[4];
      const int r0 = (wm << 5) + (lane & 15);
      const int kl = kk + ((lane >> 4) << 3);
      af[0] = mk_frag(&As[cur][r0][kl],      &As[cur][r0][kl + 16]);
      af[1] = mk_frag(&As[cur][r0 + 16][kl], &As[cur][r0 + 16][kl + 16]);
#pragma unroll
      for (int fn = 0; fn < 4; ++fn) {
        const int n0 = (wn << 6) + (fn << 4);
#pragma unroll
        for (int s = 0; s < 16; ++s)
          bfr[fn][s] = Bs[cur][kl + (s & 7) + ((s >> 3) << 4)][n0 + (lane & 15)];
      }
#pragma unroll
      for (int fm = 0; fm < 2; ++fm)
#pragma unroll
        for (int fn = 0; fn < 4; ++fn)
          acc[fm][fn] = __builtin_amdgcn_wmma_f32_16x16x32_bf16(
              false, af[fm], false, bfr[fn], (short)0, acc[fm][fn], false, false);
    }
    cur ^= 1;
  }

  // ---- epilogue: C layout VGPR r -> M = r + 8*(lane/16), N = lane%16 ----
#pragma unroll
  for (int fm = 0; fm < 2; ++fm) {
#pragma unroll
    for (int fn = 0; fn < 4; ++fn) {
      const long mbase = bm + (wm << 5) + (fm << 4) + ((lane >> 4) << 3);
      const long n     = bn + (wn << 6) + (fn << 4) + (lane & 15);
      const float bv = BIAS ? bias[n] : 0.f;
#pragma unroll
      for (int r = 0; r < 8; ++r) {
        const long m = mbase + r;
        float v = acc[fm][fn][r] + bv;
        if (RES)  v += res[m * (long)N + n];
        if (RELU) v = fmaxf(v, 0.f);
        if (OUTBF) Cb[m * (long)N + n] = f2bf(v);
        else       Cf[m * (long)N + n] = v;
      }
    }
  }
}

template<bool BIAS, bool RELU, bool RES, bool OUTBF>
static inline void gemm_go(hipStream_t st, const __bf16* A, const __bf16* Bw,
                           const float* bias, const float* res,
                           float* Cf, __bf16* Cb, int M, int N, int K) {
  dim3 g((unsigned)(N / 128), (unsigned)(M / 128));
  k_gemm<BIAS, RELU, RES, OUTBF><<<g, dim3(256), 0, st>>>(A, Bw, bias, res, Cf, Cb, M, N, K);
}

// ======================================================================
// small kernels
// ======================================================================
__global__ void k_cvt(const float* __restrict__ in, __bf16* __restrict__ out, long n) {
  long i = (long)blockIdx.x * blockDim.x + threadIdx.x;
  if (i < n) out[i] = f2bf(in[i]);
}

__global__ void k_qkvw(const float* __restrict__ Wq, const float* __restrict__ Wk,
                       const float* __restrict__ Wv, __bf16* __restrict__ out) {
  long i = (long)blockIdx.x * blockDim.x + threadIdx.x;
  const long tot = 3L * DM * DM;
  if (i >= tot) return;
  int  sect = (int)(i / ((long)DM * DM));
  long rem  = i % ((long)DM * DM);
  int  r = (int)(rem / DM), c = (int)(rem % DM);
  const float* W = (sect == 0) ? Wq : (sect == 1) ? Wk : Wv;
  out[(long)r * (3 * DM) + sect * DM + c] = f2bf(W[(long)r * DM + c]);
}

__global__ void k_qkvbias(const float* __restrict__ bq, const float* __restrict__ bk,
                          const float* __restrict__ bv, float* __restrict__ out) {
  int i = blockIdx.x * 256 + threadIdx.x;
  if (i >= 3 * DM) return;
  out[i] = (i < DM) ? bq[i] : (i < 2 * DM) ? bk[i - DM] : bv[i - 2 * DM];
}

// Wout [512,64] -> zero-padded bf16 [512,128]
__global__ void k_woutpad(const float* __restrict__ W, __bf16* __restrict__ out) {
  int i = blockIdx.x * 256 + threadIdx.x;
  if (i >= DM * 128) return;
  int r = i >> 7, c = i & 127;
  out[i] = (c < OUTD) ? f2bf(W[r * OUTD + c]) : f2bf(0.f);
}

// d_out[b,0:64) = t3[b,0:64) + bout
__global__ void k_outcopy(const float* __restrict__ t3, const float* __restrict__ bout,
                          float* __restrict__ dout) {
  int i = blockIdx.x * 256 + threadIdx.x;
  if (i >= NB * OUTD) return;
  int b = i / OUTD, d = i % OUTD;
  dout[i] = t3[(long)b * 128 + d] + bout[d];
}

// attention: one block per (b,h); S=24, HD=64
__global__ __launch_bounds__(128) void k_attn(const __bf16* __restrict__ qkv,
                                              __bf16* __restrict__ o) {
  const int b = blockIdx.x % NB;
  const int h = blockIdx.x / NB;
  __shared__ float Q[S_][HDm], Kk[S_][HDm], V[S_][HDm], Aw[S_][S_ + 1];
  for (int i = threadIdx.x; i < S_ * HDm; i += 128) {
    const int s = i / HDm, d = i % HDm;
    const __bf16* base = qkv + ((long)s * NB + b) * (3 * DM) + h * HDm + d;
    Q[s][d]  = bf2f(base[0]);
    Kk[s][d] = bf2f(base[DM]);
    V[s][d]  = bf2f(base[2 * DM]);
  }
  __syncthreads();
  for (int i = threadIdx.x; i < S_ * S_; i += 128) {
    const int s = i / S_, t = i % S_;
    float a = 0.f;
#pragma unroll 8
    for (int d = 0; d < HDm; ++d) a += Q[s][d] * Kk[t][d];
    Aw[s][t] = a * ASCALE;
  }
  __syncthreads();
  if (threadIdx.x < S_) {
    const int s = threadIdx.x;
    float m = Aw[s][0];
    for (int t = 1; t < S_; ++t) m = fmaxf(m, Aw[s][t]);
    float sum = 0.f;
    for (int t = 0; t < S_; ++t) { float e = __expf(Aw[s][t] - m); Aw[s][t] = e; sum += e; }
    float inv = 1.f / sum;
    for (int t = 0; t < S_; ++t) Aw[s][t] *= inv;
  }
  __syncthreads();
  for (int i = threadIdx.x; i < S_ * HDm; i += 128) {
    const int s = i / HDm, d = i % HDm;
    float a = 0.f;
    for (int t = 0; t < S_; ++t) a += Aw[s][t] * V[t][d];
    o[((long)s * NB + b) * DM + h * HDm + d] = f2bf(a);
  }
}

// LayerNorm over D=512; writes f32 + bf16 copies
__global__ __launch_bounds__(256) void k_ln(const float* __restrict__ in,
                                            const float* __restrict__ g,
                                            const float* __restrict__ b,
                                            float* __restrict__ outf,
                                            __bf16* __restrict__ outb) {
  __shared__ float red[256];
  const long row = blockIdx.x;
  const float* x = in + row * DM;
  const int t = threadIdx.x;
  float s = 0.f;
  for (int i = t; i < DM; i += 256) s += x[i];
  red[t] = s; __syncthreads();
  for (int o = 128; o > 0; o >>= 1) { if (t < o) red[t] += red[t + o]; __syncthreads(); }
  const float mean = red[0] / (float)DM;
  __syncthreads();
  float v = 0.f;
  for (int i = t; i < DM; i += 256) { float d = x[i] - mean; v += d * d; }
  red[t] = v; __syncthreads();
  for (int o = 128; o > 0; o >>= 1) { if (t < o) red[t] += red[t + o]; __syncthreads(); }
  const float inv = rsqrtf(red[0] / (float)DM + 1e-5f);
  for (int i = t; i < DM; i += 256) {
    float y = (x[i] - mean) * inv * g[i] + b[i];
    outf[row * DM + i] = y;
    outb[row * DM + i] = f2bf(y);
  }
}

// gate_obs = concat(cat_obs[0], cat_obs[1]) -> bf16 [NB, 2D]
__global__ void k_gateA(const float* __restrict__ cf, __bf16* __restrict__ out) {
  long i = (long)blockIdx.x * 256 + threadIdx.x;
  if (i >= (long)NB * 2 * DM) return;
  int b = (int)(i / (2 * DM)), c = (int)(i % (2 * DM));
  float v = (c < DM) ? cf[(long)b * DM + c] : cf[(long)(NB + b) * DM + (c - DM)];
  out[i] = f2bf(v);
}

// logits = gateH @ g2W + g2b  (N=5, tiny -> VALU)
__global__ void k_gate2(const __bf16* __restrict__ gh, const float* __restrict__ g2W,
                        const float* __restrict__ g2b, float* __restrict__ logits) {
  int idx = blockIdx.x * 256 + threadIdx.x;
  if (idx >= NB * (NE + 1)) return;
  int b = idx / (NE + 1), j = idx % (NE + 1);
  float acc = g2b[j];
  for (int k = 0; k < 256; ++k) acc += bf2f(gh[(long)b * 256 + k]) * g2W[k * (NE + 1) + j];
  logits[idx] = acc;
}

// bitonic sort of velocity[2048] + 4 linear-interp quantiles (0.2/0.4/0.6/0.8)
__global__ __launch_bounds__(1024) void k_quant(const float* __restrict__ vel,
                                                float* __restrict__ bins) {
  __shared__ float a[NB];
  const int t = threadIdx.x;
  a[t] = vel[t]; a[t + 1024] = vel[t + 1024];
  __syncthreads();
  for (int k = 2; k <= NB; k <<= 1)
    for (int j = k >> 1; j > 0; j >>= 1) {
      for (int base = 0; base < NB; base += 1024) {
        int i = base + t;
        int ixj = i ^ j;
        if (ixj > i) {
          bool up = ((i & k) == 0);
          float x = a[i], y = a[ixj];
          if ((x > y) == up) { a[i] = y; a[ixj] = x; }
        }
      }
      __syncthreads();
    }
  if (t < NE) {
    float q = 0.2f * (float)(t + 1) * (float)(NB - 1);
    int lo = (int)floorf(q);
    float fr = q - (float)lo;
    bins[t] = a[lo] * (1.f - fr) + a[lo + 1] * fr;
  }
}

// softmax/argmax/coef/loss per batch element
__global__ void k_gatefin(const float* __restrict__ logits, const float* __restrict__ vel,
                          const float* __restrict__ bins, float* __restrict__ coef,
                          float* __restrict__ lossb) {
  int b = blockIdx.x * 256 + threadIdx.x;
  if (b >= NB) return;
  float l[NE + 1];
  float m = -3.0e38f; int am = 0;
  for (int j = 0; j <= NE; ++j) { l[j] = logits[b * (NE + 1) + j]; if (l[j] > m) { m = l[j]; am = j; } }
  float pj[NE + 1], sum = 0.f;
  for (int j = 0; j <= NE; ++j) { pj[j] = __expf(l[j] - m); sum += pj[j]; }
  const float inv = 1.f / sum, lg = __logf(sum);
  for (int j = 0; j <= NE; ++j) pj[j] *= inv;
  const float v = vel[b];
  int ks = 0;
  for (int j = 0; j < NE; ++j) ks += (bins[j] < v) ? 1 : 0;   // searchsorted left
  lossb[b] = -(l[ks] - m - lg);
  float suf = 0.f, w[NE];
  for (int i = NE; i >= 1; --i) { suf += pj[i]; w[i - 1] = suf; }
  for (int e = 0; e < NE; ++e) coef[e * NB + b] = (am >= e + 1) ? w[e] : 0.f;
}

__global__ void k_lossred(const float* __restrict__ lossb, float* __restrict__ out) {
  __shared__ float red[256];
  const int t = threadIdx.x;
  float s = 0.f;
  for (int i = t; i < NB; i += 256) s += lossb[i];
  red[t] = s; __syncthreads();
  for (int o = 128; o > 0; o >>= 1) { if (t < o) red[t] += red[t + o]; __syncthreads(); }
  if (t == 0) out[0] = red[0] / (float)NB;
}

// emb += coef[b] * (t2[b,:] + zerob[:])
__global__ void k_eacc(float* __restrict__ embf, const float* __restrict__ t2,
                       const float* __restrict__ zb, const float* __restrict__ cf_) {
  int i = blockIdx.x * 256 + threadIdx.x;
  if (i >= NB * DM) return;
  int b = i / DM, d = i % DM;
  embf[i] += cf_[b] * (t2[i] + zb[d]);
}

// ======================================================================
// encoder layer driver (host)
// ======================================================================
struct Bufs {
  float  *hf, *t1;
  __bf16 *hb, *qkvb, *ob, *ffb;
};
static constexpr long OFF_WO = 786432;   // 512*1536
static constexpr long OFF_W1 = 1048576;  // + 512*512
static constexpr long OFF_W2 = 2097152;  // + 512*2048
static constexpr long SET_SZ = 3145728;  // + 2048*512

static void run_layer(hipStream_t st, const Bufs& B_, const __bf16* wset, const float* bqkv,
                      const float* bo, const float* b1, const float* b2,
                      const float* ln1g, const float* ln1b,
                      const float* ln2g, const float* ln2b) {
  gemm_go<true, false, false, true >(st, B_.hb,  wset,            bqkv, nullptr, nullptr, B_.qkvb, SB, 3 * DM, DM);
  k_attn<<<dim3(NB * NH), dim3(128), 0, st>>>(B_.qkvb, B_.ob);
  gemm_go<true, false, true,  false>(st, B_.ob,  wset + OFF_WO,   bo,   B_.hf,   B_.t1,  nullptr, SB, DM, DM);
  k_ln<<<dim3(SB), dim3(256), 0, st>>>(B_.t1, ln1g, ln1b, B_.hf, B_.hb);
  gemm_go<true, true,  false, true >(st, B_.hb,  wset + OFF_W1,   b1,   nullptr, nullptr, B_.ffb, SB, FFD, DM);
  gemm_go<true, false, true,  false>(st, B_.ffb, wset + OFF_W2,   b2,   B_.hf,   B_.t1,  nullptr, SB, DM, FFD);
  k_ln<<<dim3(SB), dim3(256), 0, st>>>(B_.t1, ln2g, ln2b, B_.hf, B_.hb);
}

// ======================================================================
// launch
// ======================================================================
extern "C" void kernel_launch(void* const* d_in, const int* in_sizes, int n_in,
                              void* d_out, int out_size, void* d_ws, size_t ws_size,
                              hipStream_t stream) {
  (void)in_sizes; (void)n_in; (void)out_size; (void)ws_size;

  const float* x   = (const float*)d_in[0];
  const float* vel = (const float*)d_in[1];
  const float* Win = (const float*)d_in[2];
  const float* bin = (const float*)d_in[3];
  struct L { const float *Wq,*Wk,*Wv,*Wo,*bq,*bk,*bv,*bo,*ln1g,*ln1b,*W1,*b1,*W2,*b2,*ln2g,*ln2b; };
  auto getL = [&](int base) {
    L l;
    l.Wq  =(const float*)d_in[base+0];  l.Wk  =(const float*)d_in[base+1];
    l.Wv  =(const float*)d_in[base+2];  l.Wo  =(const float*)d_in[base+3];
    l.bq  =(const float*)d_in[base+4];  l.bk  =(const float*)d_in[base+5];
    l.bv  =(const float*)d_in[base+6];  l.bo  =(const float*)d_in[base+7];
    l.ln1g=(const float*)d_in[base+8];  l.ln1b=(const float*)d_in[base+9];
    l.W1  =(const float*)d_in[base+10]; l.b1  =(const float*)d_in[base+11];
    l.W2  =(const float*)d_in[base+12]; l.b2  =(const float*)d_in[base+13];
    l.ln2g=(const float*)d_in[base+14]; l.ln2b=(const float*)d_in[base+15];
    return l;
  };
  L fl[2] = { getL(4), getL(20) };
  L ad    = getL(36);                                   // adapters, leading dim E=4
  const float* zeroW = (const float*)d_in[52];
  const float* zerob = (const float*)d_in[53];
  const float* g1W   = (const float*)d_in[54];
  const float* g1b   = (const float*)d_in[55];
  const float* g2W   = (const float*)d_in[56];
  const float* g2b   = (const float*)d_in[57];
  const float* Wout  = (const float*)d_in[58];
  const float* bout  = (const float*)d_in[59];

  // ---- workspace carve-up (bump allocator, 256B aligned) ----
  char* p = (char*)d_ws;
  auto alloc = [&](size_t bytes) { void* r = (void*)p; p += (bytes + 255) & ~(size_t)255; return r; };

  __bf16* wsets  = (__bf16*)alloc((size_t)6 * SET_SZ * 2);   // 6 layer-sets of bf16 weights
  float*  bqkvs  = (float*) alloc((size_t)6 * 3 * DM * 4);
  __bf16* Winb   = (__bf16*)alloc((size_t)IND * DM * 2);
  __bf16* g1Wb   = (__bf16*)alloc((size_t)2 * DM * 256 * 2);
  __bf16* zeroWb = (__bf16*)alloc((size_t)NE * DM * DM * 2);
  __bf16* Woutb  = (__bf16*)alloc((size_t)DM * 128 * 2);     // zero-padded N=128

  __bf16* xb   = (__bf16*)alloc((size_t)SB * IND * 2);
  float*  cf   = (float*) alloc((size_t)SB * DM * 4);
  __bf16* cb   = (__bf16*)alloc((size_t)SB * DM * 2);
  float*  hf   = (float*) alloc((size_t)SB * DM * 4);
  __bf16* hb   = (__bf16*)alloc((size_t)SB * DM * 2);
  __bf16* qkvb = (__bf16*)alloc((size_t)SB * 3 * DM * 2);
  __bf16* ob   = (__bf16*)alloc((size_t)SB * DM * 2);
  __bf16* ffb  = (__bf16*)alloc((size_t)SB * FFD * 2);
  float*  t1   = (float*) alloc((size_t)SB * DM * 4);
  __bf16* gA   = (__bf16*)alloc((size_t)NB * 2 * DM * 2);
  __bf16* gH   = (__bf16*)alloc((size_t)NB * 256 * 2);
  float*  lgts = (float*) alloc((size_t)NB * (NE + 1) * 4);
  float*  bins = (float*) alloc((size_t)NE * 4);
  float*  lssb = (float*) alloc((size_t)NB * 4);
  float*  coef = (float*) alloc((size_t)NE * NB * 4);
  float*  embf = (float*) alloc((size_t)NB * DM * 4);
  __bf16* embb = (__bf16*)alloc((size_t)NB * DM * 2);
  float*  t2   = (float*) alloc((size_t)NB * DM * 4);
  float*  t3   = (float*) alloc((size_t)NB * 128 * 4);

  auto cvt = [&](const float* in, __bf16* out, long n) {
    k_cvt<<<dim3((unsigned)((n + 255) / 256)), dim3(256), 0, stream>>>(in, out, n);
  };
  auto buildSet = [&](const float* Wq, const float* Wk, const float* Wv,
                      const float* Wo, const float* W1, const float* W2,
                      const float* bq, const float* bk, const float* bv,
                      __bf16* wb, float* bb) {
    k_qkvw<<<dim3((unsigned)((3L * DM * DM + 255) / 256)), dim3(256), 0, stream>>>(Wq, Wk, Wv, wb);
    k_qkvbias<<<dim3((3 * DM + 255) / 256), dim3(256), 0, stream>>>(bq, bk, bv, bb);
    cvt(Wo, wb + OFF_WO, (long)DM * DM);
    cvt(W1, wb + OFF_W1, (long)DM * FFD);
    cvt(W2, wb + OFF_W2, (long)FFD * DM);
  };

  // ---- weights -> bf16 ----
  for (int i = 0; i < 2; ++i)
    buildSet(fl[i].Wq, fl[i].Wk, fl[i].Wv, fl[i].Wo, fl[i].W1, fl[i].W2,
             fl[i].bq, fl[i].bk, fl[i].bv, wsets + (long)i * SET_SZ, bqkvs + (long)i * 3 * DM);
  for (int e = 0; e < NE; ++e)
    buildSet(ad.Wq + (long)e * DM * DM, ad.Wk + (long)e * DM * DM, ad.Wv + (long)e * DM * DM,
             ad.Wo + (long)e * DM * DM, ad.W1 + (long)e * DM * FFD, ad.W2 + (long)e * FFD * DM,
             ad.bq + (long)e * DM, ad.bk + (long)e * DM, ad.bv + (long)e * DM,
             wsets + (long)(2 + e) * SET_SZ, bqkvs + (long)(2 + e) * 3 * DM);
  cvt(Win,   Winb,   (long)IND * DM);
  cvt(g1W,   g1Wb,   (long)2 * DM * 256);
  cvt(zeroW, zeroWb, (long)NE * DM * DM);
  k_woutpad<<<dim3((DM * 128 + 255) / 256), dim3(256), 0, stream>>>(Wout, Woutb);

  // ---- input projection: cat_obs = x @ Win + bin ----
  cvt(x, xb, (long)SB * IND);
  gemm_go<true, false, false, false>(stream, xb, Winb, bin, nullptr, cf, nullptr, SB, DM, IND);
  cvt(cf, cb, (long)SB * DM);

  // ---- gate path ----
  k_gateA<<<dim3((unsigned)(((long)NB * 2 * DM + 255) / 256)), dim3(256), 0, stream>>>(cf, gA);
  gemm_go<true, true, false, true>(stream, gA, g1Wb, g1b, nullptr, nullptr, gH, NB, 256, 2 * DM);
  k_gate2<<<dim3((NB * (NE + 1) + 255) / 256), dim3(256), 0, stream>>>(gH, g2W, g2b, lgts);
  k_quant<<<dim3(1), dim3(1024), 0, stream>>>(vel, bins);
  k_gatefin<<<dim3((NB + 255) / 256), dim3(256), 0, stream>>>(lgts, vel, bins, coef, lssb);
  k_lossred<<<dim3(1), dim3(256), 0, stream>>>(lssb, (float*)d_out + (long)NB * OUTD);

  // ---- frozen transformer ----
  hipMemcpyAsync(hf, cf, (size_t)SB * DM * 4, hipMemcpyDeviceToDevice, stream);
  hipMemcpyAsync(hb, cb, (size_t)SB * DM * 2, hipMemcpyDeviceToDevice, stream);
  Bufs bufs{hf, t1, hb, qkvb, ob, ffb};
  for (int i = 0; i < 2; ++i)
    run_layer(stream, bufs, wsets + (long)i * SET_SZ, bqkvs + (long)i * 3 * DM,
              fl[i].bo, fl[i].b1, fl[i].b2, fl[i].ln1g, fl[i].ln1b, fl[i].ln2g, fl[i].ln2b);
  // emb = h[0] : token-0 rows are the first NB rows (seq-major layout)
  hipMemcpyAsync(embf, hf, (size_t)NB * DM * 4, hipMemcpyDeviceToDevice, stream);

  // ---- adapter experts (residual MoE) ----
  for (int e = 0; e < NE; ++e) {
    hipMemcpyAsync(hf, cf, (size_t)SB * DM * 4, hipMemcpyDeviceToDevice, stream);
    hipMemcpyAsync(hb, cb, (size_t)SB * DM * 2, hipMemcpyDeviceToDevice, stream);
    run_layer(stream, bufs, wsets + (long)(2 + e) * SET_SZ, bqkvs + (long)(2 + e) * 3 * DM,
              ad.bo + (long)e * DM, ad.b1 + (long)e * FFD, ad.b2 + (long)e * DM,
              ad.ln1g + (long)e * DM, ad.ln1b + (long)e * DM,
              ad.ln2g + (long)e * DM, ad.ln2b + (long)e * DM);
    // tok0_e @ zeroW[e]  (tok0 = first NB rows of hb)
    gemm_go<false, false, false, false>(stream, hb, zeroWb + (long)e * DM * DM,
                                        nullptr, nullptr, t2, nullptr, NB, DM, DM);
    k_eacc<<<dim3((NB * DM + 255) / 256), dim3(256), 0, stream>>>(embf, t2,
        zerob + (long)e * DM, coef + (long)e * NB);
  }

  // ---- output head: emb @ Wout(padded N=128) -> t3, then +bout into d_out ----
  cvt(embf, embb, (long)NB * DM);
  gemm_go<false, false, false, false>(stream, embb, Woutb, nullptr, nullptr,
                                      t3, nullptr, NB, 128, DM);
  k_outcopy<<<dim3((NB * OUTD + 255) / 256), dim3(256), 0, stream>>>(t3, bout, (float*)d_out);
}
